// IDAttnProcessor2_0_37529424233402
// MI455X (gfx1250) — matrix operand order repaired and checked by reading, compiled, not run on hardware
//
#include <hip/hip_runtime.h>
#include <hip/hip_bf16.h>
#include <stdint.h>

typedef __attribute__((ext_vector_type(16))) __bf16 v16bf;
typedef __attribute__((ext_vector_type(8)))  float  v8f;

union BF16Frag {
  v16bf v;
  uint4 q[2];
};

// ---- problem constants ----
#define CB    4       // batch
#define CS    4096    // seq
#define CC    1280    // channels
#define CX    2048    // cross dim
#define CT    77      // encoder tokens
#define CLID  32      // id tokens
#define CH    20      // heads
#define CD    64      // head dim
#define CTPAD 96      // padded key len for cross V (3 * 32)

__device__ __forceinline__ v8f wmma_bf16(v16bf a, v16bf b, v8f c) {
  // D = A(16x32 bf16) * B(32x16 bf16) + C(16x16 f32)
  return __builtin_amdgcn_wmma_f32_16x16x32_bf16(
      /*neg_a=*/false, a, /*neg_b=*/false, b,
      /*c_mod=*/(short)0, c, /*reuse_a=*/false, /*reuse_b=*/false);
}

// A-fragment 16x32 bf16 per ISA layout:
// lanes 0-15 (m=lane):   slots0-7 = K[k0..k0+7],   slots8-15 = K[k0+16..k0+23]
// lanes16-31 (m=lane-16): slots0-7 = K[k0+8..+15], slots8-15 = K[k0+24..+31]
__device__ __forceinline__ v16bf load_a_frag(const __bf16* A, long lda, int row0,
                                             int M, int k0, int lane) {
  int m = row0 + (lane & 15);
  if (m >= M) m = M - 1;  // clamp; garbage rows never stored
  int koff = (lane < 16) ? 0 : 8;
  const __bf16* p = A + (long)m * lda + k0 + koff;
  BF16Frag f;
  f.q[0] = *reinterpret_cast<const uint4*>(p);
  f.q[1] = *reinterpret_cast<const uint4*>(p + 16);
  return f.v;
}

// B-fragment 32x16 bf16: b[k,n] = Wrow[n][k].
// lanes 0-15 (n=lane):    slots0-7 = K[k0..k0+7],   slots8-15 = K[k0+16..k0+23]
// lanes16-31 (n=lane-16): slots0-7 = K[k0+8..+15], slots8-15 = K[k0+24..+31]
__device__ __forceinline__ v16bf load_b_frag(const __bf16* Wrow, long ldw,
                                             int n0, int k0, int lane) {
  int n = n0 + (lane & 15);
  int koff = (lane < 16) ? 0 : 8;
  const __bf16* p = Wrow + (long)n * ldw + k0 + koff;
  BF16Frag f;
  f.q[0] = reinterpret_cast<const uint4*>(p)[0];
  f.q[1] = *reinterpret_cast<const uint4*>(p + 16);
  return f.v;
}

// ---- f32 -> bf16 conversion ----
__global__ void cvt_bf16(const float* __restrict__ in, __bf16* __restrict__ out, long n) {
  long i = (long)blockIdx.x * blockDim.x + threadIdx.x;
  long stride = (long)gridDim.x * blockDim.x;
  for (; i < n; i += stride) out[i] = (__bf16)in[i];
}

// ---- GEMM: C[M,N] = A[M,K] * W[N,K]^T  (bf16 in, bf16 out) ----
__global__ __launch_bounds__(32)
void gemm_nt_bf16out(const __bf16* __restrict__ A, const __bf16* __restrict__ W,
                     __bf16* __restrict__ Cp, int M, int N, int K) {
  const int lane = threadIdx.x;
  const int row0 = blockIdx.x * 16;
  const int n0 = blockIdx.y * 64;
  v8f acc[4] = {};
  for (int k0 = 0; k0 < K; k0 += 32) {
    v16bf a = load_a_frag(A, K, row0, M, k0, lane);
#pragma unroll
    for (int j = 0; j < 4; ++j)
      acc[j] = wmma_bf16(a, load_b_frag(W, K, n0 + 16 * j, k0, lane), acc[j]);
  }
  const int mb = (lane < 16) ? 0 : 8;
  const int nn = lane & 15;
#pragma unroll
  for (int j = 0; j < 4; ++j)
#pragma unroll
    for (int r = 0; r < 8; ++r) {
      int row = row0 + mb + r;
      if (row < M) Cp[(long)row * N + n0 + 16 * j + nn] = (__bf16)acc[j][r];
    }
}

// ---- GEMM with head-transposed store: VT[((b*H+h)*64+d)*TPAD + t] ----
__global__ __launch_bounds__(32)
void gemm_nt_bf16out_vt(const __bf16* __restrict__ A, const __bf16* __restrict__ W,
                        __bf16* __restrict__ VT, int M, int N, int K,
                        int Trows, int TPAD) {
  const int lane = threadIdx.x;
  const int row0 = blockIdx.x * 16;
  const int n0 = blockIdx.y * 64;
  v8f acc[4] = {};
  for (int k0 = 0; k0 < K; k0 += 32) {
    v16bf a = load_a_frag(A, K, row0, M, k0, lane);
#pragma unroll
    for (int j = 0; j < 4; ++j)
      acc[j] = wmma_bf16(a, load_b_frag(W, K, n0 + 16 * j, k0, lane), acc[j]);
  }
  const int mb = (lane < 16) ? 0 : 8;
  const int nn = lane & 15;
#pragma unroll
  for (int j = 0; j < 4; ++j)
#pragma unroll
    for (int r = 0; r < 8; ++r) {
      int row = row0 + mb + r;
      if (row < M) {
        int b = row / Trows, t = row - b * Trows;
        int gn = n0 + 16 * j + nn;
        int h = gn >> 6, d = gn & 63;
        VT[((long)(b * CH + h) * 64 + d) * TPAD + t] = (__bf16)acc[j][r];
      }
    }
}

// ---- final GEMM: out = A*W^T + bias + residual (f32 out) ----
__global__ __launch_bounds__(32)
void gemm_nt_f32out_bias_res(const __bf16* __restrict__ A, const __bf16* __restrict__ W,
                             const float* __restrict__ bias, const float* __restrict__ resid,
                             float* __restrict__ out, int M, int N, int K) {
  const int lane = threadIdx.x;
  const int row0 = blockIdx.x * 16;
  const int n0 = blockIdx.y * 64;
  v8f acc[4] = {};
  for (int k0 = 0; k0 < K; k0 += 32) {
    v16bf a = load_a_frag(A, K, row0, M, k0, lane);
#pragma unroll
    for (int j = 0; j < 4; ++j)
      acc[j] = wmma_bf16(a, load_b_frag(W, K, n0 + 16 * j, k0, lane), acc[j]);
  }
  const int mb = (lane < 16) ? 0 : 8;
  const int nn = lane & 15;
#pragma unroll
  for (int j = 0; j < 4; ++j)
#pragma unroll
    for (int r = 0; r < 8; ++r) {
      int row = row0 + mb + r;
      int col = n0 + 16 * j + nn;
      if (row < M) {
        long idx = (long)row * N + col;
        out[idx] = acc[j][r] + bias[col] + resid[idx];
      }
    }
}

// ---- one SDPA pass (scores -> softmax -> P*V accumulate) ----
__device__ __forceinline__ void attn_part(
    v16bf aQ0, v16bf aQ1,
    const __bf16* Kh, int ldk, int nkeys, int nkt,   // nkt tiles of 16 keys
    const __bf16* Vth, int ldvt, int nkk,            // nkk k-steps of 32 for P*V
    float* sc, __bf16* pr, v8f* oc, int lane) {
  const int nn = lane & 15;
  const int mb = (lane < 16) ? 0 : 8;
  // scores: S = (Q K^T) * 0.125, masked past nkeys
  for (int kt = 0; kt < nkt; ++kt) {
    int t = kt * 16 + nn;
    int tc = (t < nkeys) ? t : (nkeys - 1);
    int koff = (lane < 16) ? 0 : 8;
    const __bf16* p = Kh + (long)tc * ldk + koff;
    BF16Frag b0, b1;
    b0.q[0] = reinterpret_cast<const uint4*>(p)[0];
    b0.q[1] = *reinterpret_cast<const uint4*>(p + 16);
    b1.q[0] = *reinterpret_cast<const uint4*>(p + 32);
    b1.q[1] = *reinterpret_cast<const uint4*>(p + 48);
    v8f c = {};
    c = wmma_bf16(aQ0, b0.v, c);
    c = wmma_bf16(aQ1, b1.v, c);
#pragma unroll
    for (int r = 0; r < 8; ++r) {
      float s = c[r] * 0.125f;
      sc[(mb + r) * 96 + kt * 16 + nn] = (t < nkeys) ? s : -3.0e38f;
    }
  }
  __syncthreads();
  // row softmax (16 rows handled by lanes 0..15), write bf16 probs zero-padded
  const int Wd = nkt * 16;
  const int PW = nkk * 32;
  if (lane < 16) {
    float mx = -3.0e38f;
    for (int j = 0; j < Wd; ++j) mx = fmaxf(mx, sc[lane * 96 + j]);
    float sum = 0.f;
    for (int j = 0; j < Wd; ++j) {
      float e = __expf(sc[lane * 96 + j] - mx);
      sum += e;
      sc[lane * 96 + j] = e;
    }
    float inv = 1.0f / sum;
    for (int j = 0; j < PW; ++j)
      pr[lane * 96 + j] = (j < Wd) ? (__bf16)(sc[lane * 96 + j] * inv) : (__bf16)0.0f;
  }
  __syncthreads();
  // O += P * V   (V stored head-transposed: Vth[d*ldvt + t])
  for (int kk = 0; kk < nkk; ++kk) {
    int koff = (lane < 16) ? 0 : 8;
    const __bf16* pp = pr + nn * 96 + kk * 32 + koff;
    BF16Frag aP;
    aP.q[0] = *reinterpret_cast<const uint4*>(pp);
    aP.q[1] = *reinterpret_cast<const uint4*>(pp + 16);
    int koff2 = (lane < 16) ? 0 : 8;
#pragma unroll
    for (int j = 0; j < 4; ++j) {
      int d = j * 16 + nn;
      const __bf16* pv = Vth + (long)d * ldvt + kk * 32 + koff2;
      BF16Frag bV;
      bV.q[0] = reinterpret_cast<const uint4*>(pv)[0];
      bV.q[1] = *reinterpret_cast<const uint4*>(pv + 16);
      oc[j] = wmma_bf16(aP.v, bV.v, oc[j]);
    }
  }
  __syncthreads();  // sc/pr reused by second call
}

// ---- fused dual attention: comb = sdpa(q,K,V) + sdpa(q,idK,idV) ----
__global__ __launch_bounds__(32)
void attn_kernel(const __bf16* __restrict__ Qb, const __bf16* __restrict__ Kc,
                 const __bf16* __restrict__ VT, const __bf16* __restrict__ Kid,
                 const __bf16* __restrict__ VidT, __bf16* __restrict__ comb) {
  __shared__ __align__(16) float  sc[16 * 96];
  __shared__ __align__(16) __bf16 pr[16 * 96];
  const int lane = threadIdx.x;
  const int qt = blockIdx.x, h = blockIdx.y, b = blockIdx.z;
  const int row0 = b * CS + qt * 16;
  const __bf16* Qh = Qb + (long)row0 * CC + h * CD;

  BF16Frag q0, q1;
  {
    int m = lane & 15;
    int koff = (lane < 16) ? 0 : 8;
    const __bf16* p = Qh + (long)m * CC + koff;
    q0.q[0] = *reinterpret_cast<const uint4*>(p);
    q0.q[1] = *reinterpret_cast<const uint4*>(p + 16);
    q1.q[0] = *reinterpret_cast<const uint4*>(p + 32);
    q1.q[1] = *reinterpret_cast<const uint4*>(p + 48);
  }
  v8f oc[4] = {};
  // cross attention: 77 keys -> 5 score tiles, 3 PV k-steps (padded to 96)
  attn_part(q0.v, q1.v, Kc + (long)(b * CT) * CC + h * CD, CC, CT, 5,
            VT + (long)(b * CH + h) * CD * CTPAD, CTPAD, 3, sc, pr, oc, lane);
  // id attention: 32 keys -> 2 score tiles, 1 PV k-step
  attn_part(q0.v, q1.v, Kid + (long)(b * CLID) * CC + h * CD, CC, CLID, 2,
            VidT + (long)(b * CH + h) * CD * CLID, CLID, 1, sc, pr, oc, lane);

  const int mb = (lane < 16) ? 0 : 8;
  const int nn = lane & 15;
#pragma unroll
  for (int j = 0; j < 4; ++j)
#pragma unroll
    for (int r = 0; r < 8; ++r)
      comb[(long)(row0 + mb + r) * CC + h * CD + j * 16 + nn] = (__bf16)oc[j][r];
}

extern "C" void kernel_launch(void* const* d_in, const int* in_sizes, int n_in,
                              void* d_out, int out_size, void* d_ws, size_t ws_size,
                              hipStream_t stream) {
  (void)in_sizes; (void)n_in; (void)out_size; (void)ws_size;
  const float* hs  = (const float*)d_in[0];  // [B,S,C]
  const float* enc = (const float*)d_in[1];  // [B,T,X]
  const float* ide = (const float*)d_in[2];  // [B,Lid,X]
  const float* Wq  = (const float*)d_in[3];  // [C,C]
  const float* Wk  = (const float*)d_in[4];  // [C,X]
  const float* Wv  = (const float*)d_in[5];
  const float* Wik = (const float*)d_in[6];
  const float* Wiv = (const float*)d_in[7];
  const float* Wo  = (const float*)d_in[8];  // [C,C]
  const float* bo  = (const float*)d_in[9];  // [C]
  float* out = (float*)d_out;

  char* ws = (char*)d_ws;
  size_t off = 0;
  auto alloc_bf16 = [&](long nelem) {
    __bf16* p = (__bf16*)(ws + off);
    off = (off + (size_t)nelem * 2 + 255) & ~(size_t)255;
    return p;
  };

  __bf16* hb   = alloc_bf16((long)CB * CS * CC);
  __bf16* eb   = alloc_bf16((long)CB * CT * CX);
  __bf16* ib   = alloc_bf16((long)CB * CLID * CX);
  __bf16* wqb  = alloc_bf16((long)CC * CC);
  __bf16* wkb  = alloc_bf16((long)CC * CX);
  __bf16* wvb  = alloc_bf16((long)CC * CX);
  __bf16* wikb = alloc_bf16((long)CC * CX);
  __bf16* wivb = alloc_bf16((long)CC * CX);
  __bf16* wob  = alloc_bf16((long)CC * CC);
  __bf16* qb   = alloc_bf16((long)CB * CS * CC);
  __bf16* kcb  = alloc_bf16((long)CB * CT * CC);
  __bf16* vtb  = alloc_bf16((long)CB * CH * CD * CTPAD);
  __bf16* kib  = alloc_bf16((long)CB * CLID * CC);
  __bf16* vitb = alloc_bf16((long)CB * CH * CD * CLID);
  __bf16* cb   = alloc_bf16((long)CB * CS * CC);

  auto cvt = [&](const float* src, __bf16* dst, long n) {
    int blocks = (int)((n + 255) / 256);
    if (blocks > 8192) blocks = 8192;
    cvt_bf16<<<blocks, 256, 0, stream>>>(src, dst, n);
  };
  cvt(hs,  hb,   (long)CB * CS * CC);
  cvt(enc, eb,   (long)CB * CT * CX);
  cvt(ide, ib,   (long)CB * CLID * CX);
  cvt(Wq,  wqb,  (long)CC * CC);
  cvt(Wk,  wkb,  (long)CC * CX);
  cvt(Wv,  wvb,  (long)CC * CX);
  cvt(Wik, wikb, (long)CC * CX);
  cvt(Wiv, wivb, (long)CC * CX);
  cvt(Wo,  wob,  (long)CC * CC);

  dim3 blk(32);
  const int Mq = CB * CS;        // 16384
  const int Me = CB * CT;        // 308
  const int Mi = CB * CLID;      // 128

  // Q = hidden @ Wq^T
  gemm_nt_bf16out<<<dim3(Mq / 16, CC / 64), blk, 0, stream>>>(hb, wqb, qb, Mq, CC, CC);
  // K = enc @ Wk^T ; V = enc @ Wv^T (head-transposed)
  gemm_nt_bf16out<<<dim3((Me + 15) / 16, CC / 64), blk, 0, stream>>>(eb, wkb, kcb, Me, CC, CX);
  gemm_nt_bf16out_vt<<<dim3((Me + 15) / 16, CC / 64), blk, 0, stream>>>(eb, wvb, vtb, Me, CC, CX, CT, CTPAD);
  // idK = id @ Wik^T ; idV = id @ Wiv^T (head-transposed)
  gemm_nt_bf16out<<<dim3(Mi / 16, CC / 64), blk, 0, stream>>>(ib, wikb, kib, Mi, CC, CX);
  gemm_nt_bf16out_vt<<<dim3(Mi / 16, CC / 64), blk, 0, stream>>>(ib, wivb, vitb, Mi, CC, CX, CLID, CLID);
  // combined = sdpa(q,K,V) + sdpa(q,idK,idV)
  attn_kernel<<<dim3(CS / 16, CH, CB), blk, 0, stream>>>(qb, kcb, vtb, kib, vitb, cb);
  // out = combined @ Wo^T + bo + residual
  gemm_nt_f32out_bias_res<<<dim3(Mq / 16, CC / 64), blk, 0, stream>>>(cb, wob, bo, hs, out, Mq, CC, CC);
}
